// EfficientLeaf_31516470018362
// MI455X (gfx1250) — matrix operand order repaired and checked by reading, not compiled, not run on hardware
//
#include <hip/hip_runtime.h>
#include <hip/hip_bf16.h>
#include <math.h>
#include <stdint.h>

typedef __attribute__((ext_vector_type(16))) _Float16 v16h;
typedef __attribute__((ext_vector_type(8)))  float    v8f;

#define SRLEN 160000
#define BATCH 16
#define NFILT 40
#define TTOT  1000   // output frames

// ---------------------------------------------------------------------------
// Gabor kernel generation: rows 0..9 = real, 10..19 = imag, 20..31 = zero pad.
// Layout: kern[row][k], row stride KP (K padded to multiple of 32).
// ---------------------------------------------------------------------------
template<int A0, int K, int KP>
__global__ void gen_gabor(const float* __restrict__ cf, const float* __restrict__ bw,
                          _Float16* __restrict__ kern) {
    const float SIG_LO = 0.74957460f;   // 2*sqrt(2 ln2)/pi
    const float SIG_HI = 150.28971f;    // 401*sqrt(2 ln2)/pi
    for (int e = threadIdx.x; e < 32 * KP; e += blockDim.x) {
        int row = e / KP, k = e % KP;
        float v = 0.0f;
        if (row < 20 && k < K) {
            int f = A0 + (row < 10 ? row : row - 10);
            float mu = fminf(fmaxf(cf[f], 0.0f), 3.14159265358979f);
            float sg = fminf(fmaxf(bw[f], SIG_LO), SIG_HI);
            float t  = (float)(k - K / 2);
            float g  = expf(-t * t / (2.0f * sg * sg)) * 0.3989422804f / sg;
            v = (row < 10) ? g * cosf(mu * t) : g * sinf(mu * t);
        }
        kern[e] = (_Float16)v;
    }
}

// Gaussian pooling windows: win[f][w], f local to group.
template<int A0, int CS, int W>
__global__ void gen_win(const float* __restrict__ pw, float* __restrict__ win) {
    for (int e = threadIdx.x; e < 10 * W; e += blockDim.x) {
        int f = e / W, w = e % W;
        float p  = fminf(fmaxf(pw[A0 + f], 2.0f / 401.0f), 0.5f);
        float sp = p / (float)CS * 401.0f / (float)W;
        float t2 = (float)w * (2.0f / (float)(W - 1)) - 1.0f;
        float q  = t2 / sp;
        win[e] = expf(-0.5f * q * q);
    }
}

// ---------------------------------------------------------------------------
// Fused Gabor-conv (WMMA f16->f32) + |.|^2 + Gaussian pool + log1p.
// Block: (tile of TOUT pooled frames) x (batch).  256 threads = 8 waves.
// x staged into LDS with GLOBAL_LOAD_ASYNC_TO_LDS (ASYNCcnt), then cvt to f16.
// ---------------------------------------------------------------------------
template<int A0, int CS, int PS, int K, int W, int TOUT, int NSL>
__global__ __launch_bounds__(256) void convpool(
    const float* __restrict__ x, const _Float16* __restrict__ kern,
    const float* __restrict__ win, const float* __restrict__ log1pa,
    float* __restrict__ yfull) {
    constexpr int NT1    = (TOUT - 1) * PS + W;       // conv positions needed
    constexpr int NTILES = (NT1 + 15) / 16;
    constexpr int KC     = (K + 31) / 32;             // k-chunks of 32
    constexpr int KP     = KC * 32;
    constexpr int NXA    = (NTILES * 16 - 1) * CS + KP; // x samples staged
    constexpr int T1     = SRLEN / CS;                // conv output length

    __shared__ _Float16 xs[NXA];
    // f32 staging buffer is dead after conversion; conv scratch reuses its LDS.
    __shared__ union {
        float stage[NXA];
        float conv[20 * NT1];
    } u;
    __shared__ float psum[256];

    const int b    = blockIdx.y;
    const int t0   = blockIdx.x * TOUT;
    const int tid  = threadIdx.x;
    const int lane = tid & 31;
    const int wave = tid >> 5;

    // ---- Stage x slice into LDS (f32) via async loads; zero-fill OOB. ----
    const int  t1base = t0 * PS - W / 2;
    const long xbase  = (long)t1base * CS - K / 2;
    const uint64_t xaddr = (uint64_t)(x + (size_t)b * SRLEN);
    for (int j = tid; j < NXA; j += 256) {
        long gi = xbase + j;
        if (gi >= 0 && gi < SRLEN) {
            unsigned ldsoff = (unsigned)(uintptr_t)(&u.stage[j]); // LDS byte offset
            unsigned voff   = (unsigned)gi * 4u;
            asm volatile("global_load_async_to_lds_b32 %0, %1, %2"
                         :: "v"(ldsoff), "v"(voff), "s"(xaddr) : "memory");
        } else {
            u.stage[j] = 0.0f;
        }
    }

    // ---- Preload A fragments (kernel matrix) into registers (overlaps). ----
    // 16-bit A layout: lanes 0-15 hold K {kh..kh+7} in halves 0..7 and
    // {16+kh..16+kh+7} in halves 8..15 (kh = 0 / 8 by lane group).
    v16h afrag[KC][2];
    {
        int m  = lane & 15;
        int kh = (lane >> 4) * 8;
#pragma unroll
        for (int kc = 0; kc < KC; ++kc)
#pragma unroll
            for (int mt = 0; mt < 2; ++mt) {
                const _Float16* row = kern + (size_t)(mt * 16 + m) * KP + kc * 32;
                v16h a;
#pragma unroll
                for (int h = 0; h < 8; ++h) a[h]     = row[kh + h];
#pragma unroll
                for (int h = 0; h < 8; ++h) a[8 + h] = row[16 + kh + h];
                afrag[kc][mt] = a;
            }
    }

    // Wait for this wave's async LDS writes, then sync all waves.
    asm volatile("s_wait_asynccnt 0x0" ::: "memory");
    __syncthreads();

    // ---- Convert staged f32 -> f16 operand buffer. ----
    for (int j = tid; j < NXA; j += 256) xs[j] = (_Float16)u.stage[j];
    __syncthreads();

    // ---- Conv tiles: D[ch, i] = sum_k kern[ch,k] * xs[i*CS + k]. ----
    for (int nt = wave; nt < NTILES; nt += 8) {
        v8f acc0 = {}; v8f acc1 = {};
        const int n     = lane & 15;
        const int kb    = (lane >> 4) * 8;    // B layout: K {kb..kb+7} then {16+kb..16+kb+7} per lane
        const int ibase = (nt * 16 + n) * CS + kb;
        v16h bcur;
#pragma unroll
        for (int h = 0; h < 16; ++h) bcur[h] = xs[ibase + h + (h >= 8 ? 8 : 0)];
#pragma unroll
        for (int kc = 0; kc < KC; ++kc) {
            v16h bnext = bcur;
            if (kc + 1 < KC) {   // software pipeline: next B under current WMMAs
#pragma unroll
                for (int h = 0; h < 16; ++h) bnext[h] = xs[ibase + (kc + 1) * 32 + h + (h >= 8 ? 8 : 0)];
            }
            acc0 = __builtin_amdgcn_wmma_f32_16x16x32_f16(
                false, afrag[kc][0], false, bcur, (short)0, acc0, false, false);
            acc1 = __builtin_amdgcn_wmma_f32_16x16x32_f16(
                false, afrag[kc][1], false, bcur, (short)0, acc1, false, false);
            bcur = bnext;
        }
        const int i = nt * 16 + n;
        if (i < NT1) {
            const int mrow = (lane >> 4) * 8;
#pragma unroll
            for (int r = 0; r < 8; ++r) {
                int ch0 = mrow + r;                    // 0..15
                u.conv[ch0 * NT1 + i] = acc0[r];
                int ch1 = 16 + ch0;                    // only 16..19 valid
                if (ch1 < 20) u.conv[ch1 * NT1 + i] = acc1[r];
            }
        }
    }
    __syncthreads();

    // ---- Pooling: NSL lanes cooperate per (filter, frame) output. ----
    const int outIdx = tid / NSL;
    const int sl     = tid % NSL;
    const int fl     = outIdx / TOUT;
    const int tl     = outIdx % TOUT;
    const bool valid = (outIdx < 10 * TOUT) && (t0 + tl < TTOT);
    float s = 0.0f;
    if (valid) {
        for (int w = sl; w < W; w += NSL) {
            int t1l = tl * PS + w;
            int t1g = t1base + t1l;
            if (t1g >= 0 && t1g < T1) {
                float re = u.conv[fl * NT1 + t1l];
                float im = u.conv[(10 + fl) * NT1 + t1l];
                s += win[fl * W + w] * (re * re + im * im);
            }
        }
    }
    psum[tid] = s;
    __syncthreads();
    if (valid && sl == 0) {
        float tot = 0.0f;
#pragma unroll
        for (int q = 0; q < NSL; ++q) tot += psum[outIdx * NSL + q];
        float a10 = expf(log1pa[A0 + fl] * 2.302585092994046f); // 10^a
        yfull[((size_t)b * NFILT + (A0 + fl)) * TTOT + (t0 + tl)] = log1pf(a10 * tot);
    }
}

// ---------------------------------------------------------------------------
// Exact median (rank 499 of ascending sort over T=1000) per (b, f).
// ---------------------------------------------------------------------------
__global__ void median_k(const float* __restrict__ yfull, float* __restrict__ med) {
    __shared__ float vals[TTOT];
    const int bf = blockIdx.x;
    const float* row = yfull + (size_t)bf * TTOT;
    for (int t = threadIdx.x; t < TTOT; t += blockDim.x) vals[t] = row[t];
    __syncthreads();
    for (int i = threadIdx.x; i < TTOT; i += blockDim.x) {
        float v = vals[i];
        int lo = 0, eq = 0;
        for (int j = 0; j < TTOT; ++j) {
            lo += (vals[j] < v);
            eq += (vals[j] == v);
        }
        if (lo <= (TTOT - 1) / 2 && (TTOT - 1) / 2 < lo + eq) med[bf] = v;
    }
}

// BN stats per channel c in [0,80): c<40 -> y, c>=40 -> y - med.
__global__ void bnstats_k(const float* __restrict__ yfull, const float* __restrict__ med,
                          float* __restrict__ mean, float* __restrict__ rstd) {
    __shared__ double sd[256];
    __shared__ double sq[256];
    const int c = blockIdx.x;
    const int f = c % NFILT, cc = c / NFILT;
    double s = 0.0, s2 = 0.0;
    for (int i = threadIdx.x; i < BATCH * TTOT; i += blockDim.x) {
        int b = i / TTOT, t = i % TTOT;
        float v = yfull[((size_t)b * NFILT + f) * TTOT + t];
        if (cc) v -= med[b * NFILT + f];
        s += (double)v; s2 += (double)v * (double)v;
    }
    sd[threadIdx.x] = s; sq[threadIdx.x] = s2;
    __syncthreads();
    for (int st = 128; st > 0; st >>= 1) {
        if (threadIdx.x < st) {
            sd[threadIdx.x] += sd[threadIdx.x + st];
            sq[threadIdx.x] += sq[threadIdx.x + st];
        }
        __syncthreads();
    }
    if (threadIdx.x == 0) {
        double m   = sd[0] / (double)(BATCH * TTOT);
        double var = sq[0] / (double)(BATCH * TTOT) - m * m;
        mean[c] = (float)m;
        rstd[c] = rsqrtf((float)var + 1e-5f);
    }
}

__global__ void bnapply_k(const float* __restrict__ yfull, const float* __restrict__ med,
                          const float* __restrict__ mean, const float* __restrict__ rstd,
                          const float* __restrict__ gamma, const float* __restrict__ beta,
                          float* __restrict__ out) {
    int idx = blockIdx.x * blockDim.x + threadIdx.x;
    if (idx >= BATCH * 2 * NFILT * TTOT) return;
    int t  = idx % TTOT;
    int f  = (idx / TTOT) % NFILT;
    int cc = (idx / (TTOT * NFILT)) % 2;
    int b  = idx / (TTOT * NFILT * 2);
    int c  = cc * NFILT + f;
    float v = yfull[((size_t)b * NFILT + f) * TTOT + t];
    if (cc) v -= med[b * NFILT + f];
    out[idx] = (v - mean[c]) * rstd[c] * gamma[c] + beta[c];
}

// ---------------------------------------------------------------------------
extern "C" void kernel_launch(void* const* d_in, const int* in_sizes, int n_in,
                              void* d_out, int out_size, void* d_ws, size_t ws_size,
                              hipStream_t stream) {
    (void)in_sizes; (void)n_in; (void)out_size;
    const float* x   = (const float*)d_in[0];
    const float* cf  = (const float*)d_in[1];
    const float* bw  = (const float*)d_in[2];
    const float* pw  = (const float*)d_in[3];
    const float* la  = (const float*)d_in[4];
    const float* gam = (const float*)d_in[5];
    const float* bet = (const float*)d_in[6];
    float* out = (float*)d_out;

    char* ws = (char*)d_ws;
    size_t off = 0;
    float* yfull = (float*)(ws + off); off += (size_t)BATCH * NFILT * TTOT * 4; // 2.56 MB
    float* med   = (float*)(ws + off); off += (size_t)BATCH * NFILT * 4;
    float* mean  = (float*)(ws + off); off += 80 * 4;
    float* rstd  = (float*)(ws + off); off += 80 * 4;
    off = (off + 255) & ~(size_t)255;
    _Float16* k0 = (_Float16*)(ws + off); off += 32 * 160 * 2;
    _Float16* k1 = (_Float16*)(ws + off); off += 32 * 96 * 2;
    _Float16* k2 = (_Float16*)(ws + off); off += 32 * 64 * 2;
    _Float16* k3 = (_Float16*)(ws + off); off += 32 * 32 * 2;
    off = (off + 255) & ~(size_t)255;
    float* w0 = (float*)(ws + off); off += 10 * 41 * 4;
    float* w1 = (float*)(ws + off); off += 10 * 101 * 4;
    float* w2 = (float*)(ws + off); off += 10 * 201 * 4;
    float* w3 = (float*)(ws + off); off += 10 * 401 * 4;
    if (ws_size < off) return;

    // Filter / window generation (group cfg: a, cs, ps, K, W)
    gen_gabor< 0, 151, 160><<<1, 256, 0, stream>>>(cf, bw, k0);
    gen_gabor<10,  83,  96><<<1, 256, 0, stream>>>(cf, bw, k1);
    gen_gabor<20,  47,  64><<<1, 256, 0, stream>>>(cf, bw, k2);
    gen_gabor<30,  25,  32><<<1, 256, 0, stream>>>(cf, bw, k3);
    gen_win< 0, 10,  41><<<1, 256, 0, stream>>>(pw, w0);
    gen_win<10,  4, 101><<<1, 256, 0, stream>>>(pw, w1);
    gen_win<20,  2, 201><<<1, 256, 0, stream>>>(pw, w2);
    gen_win<30,  1, 401><<<1, 256, 0, stream>>>(pw, w3);

    // Fused conv(WMMA) + pool + log1p, one launch per group.
    // <A0, CS, PS, K, W, TOUT, NSL>
    dim3 g0((TTOT + 15) / 16, BATCH);
    convpool< 0, 10,  16, 151,  41, 16, 1><<<g0, 256, 0, stream>>>(x, k0, w0, la, yfull);
    dim3 g1(TTOT / 8, BATCH);
    convpool<10,  4,  40,  83, 101,  8, 2><<<g1, 256, 0, stream>>>(x, k1, w1, la, yfull);
    dim3 g2(TTOT / 4, BATCH);
    convpool<20,  2,  80,  47, 201,  4, 4><<<g2, 256, 0, stream>>>(x, k2, w2, la, yfull);
    dim3 g3(TTOT / 2, BATCH);
    convpool<30,  1, 160,  25, 401,  2, 8><<<g3, 256, 0, stream>>>(x, k3, w3, la, yfull);

    median_k<<<BATCH * NFILT, 256, 0, stream>>>(yfull, med);
    bnstats_k<<<80, 256, 0, stream>>>(yfull, med, mean, rstd);
    int total = BATCH * 2 * NFILT * TTOT;
    bnapply_k<<<(total + 255) / 256, 256, 0, stream>>>(yfull, med, mean, rstd, gam, bet, out);
}
